// SpikingLanguageCore_24730421690976
// MI455X (gfx1250) — matrix operand order repaired and checked by reading, not compiled, not run on hardware
//
#include <hip/hip_runtime.h>
#include <hip/hip_bf16.h>

typedef __attribute__((ext_vector_type(16))) __bf16 v16bf;
typedef __attribute__((ext_vector_type(8)))  float  v8f;

#define B_SZ 16384
#define H_SZ 1024
#define T_STEPS 20
#define LEAK_F 0.05f          // DT/TAU = 0.001/0.02
#define KCH 32                // H / 32 K-chunks per WMMA k-loop
#define TILE_ELEMS 512        // 32 lanes * 16 bf16 per (tile, kchunk) operand block

// ---------------------------------------------------------------------------
// Pack an f32 matrix into bf16 hi/lo operand tiles in the CDNA5 WMMA register
// striping, so the GEMM inner loop does pure coalesced b128 loads.
//   A layout (16-bit A 16x32): lane<16 -> M=lane, K = {0..7, 16..23}
//                              lane>=16 -> M=lane-16, K = {8..15, 24..31}
//   B layout (16-bit B 32x16): lane<16 -> N=lane, K = {0..7, 16..23}
//                              lane>=16 -> N=lane-16, K = {8..15, 24..31}
// element(outer, k) = src[outer*outerStride + k*kStride]
// ---------------------------------------------------------------------------
__global__ __launch_bounds__(256) void pack_split_kernel(
    const float* __restrict__ src, __bf16* __restrict__ dhi, __bf16* __restrict__ dlo,
    long long outerStride, long long kStride, int layoutB, long long total)
{
  long long e = (long long)blockIdx.x * 256 + threadIdx.x;
  if (e >= total) return;
  int pos  = (int)(e & 15);
  int lane = (int)((e >> 4) & 31);
  int kc   = (int)((e >> 9) & (KCH - 1));
  long long ot = e >> 14;                       // operand tile (16 rows/cols)
  int outer = (int)(ot << 4) + (lane & 15);
  int hb = lane >> 4;
  int k32 = layoutB ? ((((pos >> 3) & 1) << 4) | (hb << 3) | (pos & 7))
                    : ((((pos >> 3) & 1) << 4) | (hb << 3) | (pos & 7));
  long long k = ((long long)kc << 5) | k32;
  float x = src[(long long)outer * outerStride + k * kStride];
  __bf16 h = (__bf16)x;
  dhi[e] = h;
  dlo[e] = (__bf16)(x - (float)h);              // residual for split-bf16 GEMM
}

// ---------------------------------------------------------------------------
// Fused SNN core: one workgroup owns 32 batch rows (2 M-tiles) for all 20
// timesteps. 512 threads = 16 waves; wave w computes C tiles
// {M-tile 0,1} x {N-tile 4w..4w+3}  ->  each B (hi/lo) register load feeds
// 2 WMMAs (halves the dominant L2 Wr stream vs 1 M-tile blocking).
//
// LDS (288 KB of the 320 KB WGP pool):
//   [0    , 128K) v        f32 32x1024
//   [128K , 160K) spikesum u8  32x1024   (count <= 20)
//   [160K , 224K) spike ping (bf16, A-operand-packed, 2 M-tiles)
//   [224K , 288K) spike pong
// proj lives in registers (fixed owner lane per element across all steps).
// ---------------------------------------------------------------------------
__global__ __launch_bounds__(512) void snn_core_kernel(
    const __bf16* __restrict__ inAh, const __bf16* __restrict__ inAl,
    const __bf16* __restrict__ WpBh, const __bf16* __restrict__ WpBl,
    const __bf16* __restrict__ WrBh, const __bf16* __restrict__ WrBl,
    const float* __restrict__ prev_v, const float* __restrict__ prev_s,
    const float* __restrict__ bp, float* __restrict__ out)
{
  extern __shared__ char lds[];
  float*         v_l   = (float*)(lds + 0);
  unsigned char* sum_l = (unsigned char*)(lds + 131072);
  __bf16*        spkA  = (__bf16*)(lds + 163840);
  __bf16*        spkB  = (__bf16*)(lds + 229376);

  const int tid  = threadIdx.x;
  const int lane = tid & 31;
  const int wave = tid >> 5;                       // 0..15
  const long long gbase = (long long)blockIdx.x * 32 * H_SZ;

  // init voltage / spike-count, and pack prev_spike into A-operand layout
  for (int i = tid; i < 32 * H_SZ; i += 512) {
    v_l[i] = prev_v[gbase + i];
    sum_l[i] = 0;
    float sp = prev_s[gbase + i];
    int m = i >> 10, k = i & (H_SZ - 1);           // m: 0..31
    int mt = m >> 4, ml = m & 15;
    int kc = k >> 5, k32 = k & 31;
    int pl = ml | (((k32 >> 3) & 1) << 4);
    int pp = (k32 & 7) | (((k32 >> 4) & 1) << 3);
    spkA[(mt << 14) | (kc << 9) | (pl << 4) | pp] = (__bf16)sp;
  }

  const v8f zero = {0.f,0.f,0.f,0.f,0.f,0.f,0.f,0.f};
  v8f acc[2][4];
  v8f proj[2][4];

  // ---- proj = input @ Wp + bp : 3-term split-bf16 GEMM (one time) ----
  {
#pragma unroll
    for (int mt = 0; mt < 2; ++mt)
#pragma unroll
      for (int t = 0; t < 4; ++t) acc[mt][t] = zero;
    for (int kc = 0; kc < KCH; ++kc) {
      v16bf ah[2], al[2];
#pragma unroll
      for (int mt = 0; mt < 2; ++mt) {
        long long aoff = ((long long)(blockIdx.x * 2 + mt) * KCH + kc) * TILE_ELEMS + lane * 16;
        ah[mt] = *(const v16bf*)(inAh + aoff);
        al[mt] = *(const v16bf*)(inAl + aoff);
      }
#pragma unroll
      for (int t = 0; t < 4; ++t) {
        long long boff = ((long long)(wave * 4 + t) * KCH + kc) * TILE_ELEMS + lane * 16;
        v16bf bh = *(const v16bf*)(WpBh + boff);
        v16bf bl = *(const v16bf*)(WpBl + boff);
#pragma unroll
        for (int mt = 0; mt < 2; ++mt) {
          acc[mt][t] = __builtin_amdgcn_wmma_f32_16x16x32_bf16(false, ah[mt], false, bh, (short)0, acc[mt][t], false, false);
          acc[mt][t] = __builtin_amdgcn_wmma_f32_16x16x32_bf16(false, ah[mt], false, bl, (short)0, acc[mt][t], false, false);
          acc[mt][t] = __builtin_amdgcn_wmma_f32_16x16x32_bf16(false, al[mt], false, bh, (short)0, acc[mt][t], false, false);
        }
      }
    }
    const int nl = lane & 15;
#pragma unroll
    for (int mt = 0; mt < 2; ++mt)
#pragma unroll
      for (int t = 0; t < 4; ++t) {
        float bb = bp[(wave * 4 + t) * 16 + nl];
#pragma unroll
        for (int r = 0; r < 8; ++r) proj[mt][t][r] = acc[mt][t][r] + bb;
      }
  }
  __syncthreads();

  // ---- 20 leaky-integrate-fire steps; rec = s @ Wr (s exact in bf16,
  //      Wr split hi/lo -> 2 WMMAs per tile for ~fp32 fidelity) ----
  for (int step = 0; step < T_STEPS; ++step) {
    const __bf16* rd = (step & 1) ? spkB : spkA;
    __bf16*       wr = (step & 1) ? spkA : spkB;
#pragma unroll
    for (int mt = 0; mt < 2; ++mt)
#pragma unroll
      for (int t = 0; t < 4; ++t) acc[mt][t] = zero;

    for (int kc = 0; kc < KCH; ++kc) {
      v16bf a[2];
#pragma unroll
      for (int mt = 0; mt < 2; ++mt)
        a[mt] = *(const v16bf*)(rd + (mt << 14) + (kc << 9) + (lane << 4));
#pragma unroll
      for (int t = 0; t < 4; ++t) {
        long long boff = ((long long)(wave * 4 + t) * KCH + kc) * TILE_ELEMS + lane * 16;
        v16bf bh = *(const v16bf*)(WrBh + boff);
        v16bf bl = *(const v16bf*)(WrBl + boff);
#pragma unroll
        for (int mt = 0; mt < 2; ++mt) {
          acc[mt][t] = __builtin_amdgcn_wmma_f32_16x16x32_bf16(false, a[mt], false, bh, (short)0, acc[mt][t], false, false);
          acc[mt][t] = __builtin_amdgcn_wmma_f32_16x16x32_bf16(false, a[mt], false, bl, (short)0, acc[mt][t], false, false);
        }
      }
    }

    const int mAdd = (lane >> 4) << 3;             // C/D layout: lanes 16-31 hold M+8
    const int nl = lane & 15;
    const bool last = (step == T_STEPS - 1);
#pragma unroll
    for (int mt = 0; mt < 2; ++mt)
#pragma unroll
      for (int t = 0; t < 4; ++t) {
        int n = (wave * 4 + t) * 16 + nl;
#pragma unroll
        for (int r = 0; r < 8; ++r) {
          int ml = r + mAdd;                       // 0..15 within M-tile
          int m = (mt << 4) | ml;                  // 0..31 within workgroup
          int idx = m * H_SZ + n;
          float pv = v_l[idx];
          float vn = pv + (acc[mt][t][r] + proj[mt][t][r] - pv) * LEAK_F;
          float spike = (vn >= 1.0f) ? 1.0f : 0.0f;
          vn = vn * (1.0f - spike);                // reset (V_RESET = 0)
          v_l[idx] = vn;
          int cnt = (int)sum_l[idx] + (spike != 0.0f ? 1 : 0);
          sum_l[idx] = (unsigned char)cnt;
          // scatter spike into next step's A-operand layout
          int kc2 = n >> 5, k32 = n & 31;
          int pl = ml | (((k32 >> 3) & 1) << 4);
          int pp = (k32 & 7) | (((k32 >> 4) & 1) << 3);
          wr[(mt << 14) | (kc2 << 9) | (pl << 4) | pp] = (__bf16)spike;
          if (last) {
            long long g = gbase + idx;
            out[g] = (float)cnt * (1.0f / T_STEPS);           // avg_spikes
            out[(long long)B_SZ * H_SZ + g] = vn;             // final v
            out[2LL * B_SZ * H_SZ + g] = spike;               // final s
          }
        }
      }
    __syncthreads();
  }
}

extern "C" void kernel_launch(void* const* d_in, const int* in_sizes, int n_in,
                              void* d_out, int out_size, void* d_ws, size_t ws_size,
                              hipStream_t stream) {
  const float* input_state  = (const float*)d_in[0];
  const float* prev_voltage = (const float*)d_in[1];
  const float* prev_spike   = (const float*)d_in[2];
  const float* Wp           = (const float*)d_in[3];
  const float* bp           = (const float*)d_in[4];
  const float* Wr           = (const float*)d_in[5];

  const size_t nIn = (size_t)B_SZ * H_SZ;     // 16M elements
  const size_t nW  = (size_t)H_SZ * H_SZ;     // 1M elements
  __bf16* inAh = (__bf16*)d_ws;               // ws usage: 2*32MB + 2*2MB + 2*2MB = 72 MB
  __bf16* inAl = inAh + nIn;
  __bf16* WpBh = inAl + nIn;
  __bf16* WpBl = WpBh + nW;
  __bf16* WrBh = WpBl + nW;
  __bf16* WrBl = WrBh + nW;

  // pack: input as A-operand (row-major [B,K]); Wp/Wr as B-operand ([K,N])
  pack_split_kernel<<<(int)(nIn / 256), 256, 0, stream>>>(
      input_state, inAh, inAl, (long long)H_SZ, 1LL, 0, (long long)nIn);
  pack_split_kernel<<<(int)(nW / 256), 256, 0, stream>>>(
      Wp, WpBh, WpBl, 1LL, (long long)H_SZ, 1, (long long)nW);
  pack_split_kernel<<<(int)(nW / 256), 256, 0, stream>>>(
      Wr, WrBh, WrBl, 1LL, (long long)H_SZ, 1, (long long)nW);

  // 512 workgroups x 512 threads, 288 KB dynamic LDS (of 320 KB WGP pool)
  snn_core_kernel<<<B_SZ / 32, 512, 294912, stream>>>(
      inAh, inAl, WpBh, WpBl, WrBh, WrBl,
      prev_voltage, prev_spike, bp, (float*)d_out);

  (void)in_sizes; (void)n_in; (void)out_size; (void)ws_size;
}
